// glbuilder_62792421868044
// MI455X (gfx1250) — matrix operand order repaired and checked by reading, not compiled, not run on hardware
//
#include <hip/hip_runtime.h>

#define B_   32
#define GS_  1024
#define P_   128
#define OUTW 1152            // 1024 + 128
#define LDSTRIDE 68          // u32 per LDS row: 64 data + 4 pad (272B = 17*16B: b128-aligned rows, conflict-free)

typedef __attribute__((ext_vector_type(16))) __bf16 v16bf;
typedef __attribute__((ext_vector_type(8)))  float  v8f;
typedef __attribute__((ext_vector_type(4)))  int    v4i;

union Frag { unsigned int u[8]; uint4 q[2]; v16bf v; };

static __device__ __forceinline__ unsigned short f32_to_bf16(float f) {
    unsigned int u = __float_as_uint(f);
    unsigned int r = u + 0x7FFFu + ((u >> 16) & 1u);   // round-to-nearest-even
    return (unsigned short)(r >> 16);
}
static __device__ __forceinline__ float bf16_to_f32(unsigned short h) {
    return __uint_as_float(((unsigned int)h) << 16);
}

// gfx1250 async global->LDS DMA (ASYNCcnt) if the toolchain declares it
#if defined(__gfx1250__) && __has_builtin(__builtin_amdgcn_global_load_async_to_lds_b128)
#define HAVE_ASYNC_LDS 1
// signature (probe-derived): (v4i AS1* gsrc, v4i AS3* ldst, imm offset, imm cpol)
#define GASYNC_B128(gp, lp)                                             \
    __builtin_amdgcn_global_load_async_to_lds_b128(                     \
        (__attribute__((address_space(1))) v4i*)(gp),                   \
        (__attribute__((address_space(3))) v4i*)(lp), 0, 0)
#endif

// ---------------------------------------------------------------------------
// Kernel 1: y = x @ (M + M^T) in fp32; emit bf16 hi/lo splits of x and y,
// and d = 0.5 * rowdot(y, x).  One block = 16 rows of (b,i).
// ---------------------------------------------------------------------------
__global__ void prep_kernel(const float* __restrict__ x, const float* __restrict__ metrik,
                            unsigned short* __restrict__ yh, unsigned short* __restrict__ yl,
                            unsigned short* __restrict__ xh, unsigned short* __restrict__ xl,
                            float* __restrict__ dvec)
{
    extern __shared__ char dynsmem[];
    float* S  = (float*)dynsmem;          // 128*128 f32 = 64KB
    float* xr = S  + P_ * P_;             // 16*128 f32
    float* yr = xr + 16 * P_;             // 16*128 f32

    const int  tid  = threadIdx.x;        // 256 threads
    const long row0 = (long)blockIdx.x * 16;   // flattened (b*GS + i)

    for (int idx = tid; idx < P_ * P_; idx += 256) {
        int p = idx >> 7, q = idx & 127;
        S[idx] = metrik[p * P_ + q] + metrik[q * P_ + p];
    }
    for (int idx = tid; idx < 16 * P_; idx += 256) {
        float v = x[row0 * P_ + idx];
        xr[idx] = v;
        unsigned short h = f32_to_bf16(v);
        unsigned short l = f32_to_bf16(v - bf16_to_f32(h));
        xh[row0 * P_ + idx] = h;
        xl[row0 * P_ + idx] = l;
    }
    __syncthreads();

    const int q = tid & 127;
    for (int rr = 0; rr < 8; ++rr) {
        int r = rr * 2 + (tid >> 7);
        const float* xp = xr + r * P_;
        float acc = 0.f;
        #pragma unroll 8
        for (int p = 0; p < P_; ++p) acc = fmaf(xp[p], S[p * P_ + q], acc);
        yr[r * P_ + q] = acc;
        unsigned short h = f32_to_bf16(acc);
        unsigned short l = f32_to_bf16(acc - bf16_to_f32(h));
        yh[(row0 + r) * P_ + q] = h;
        yl[(row0 + r) * P_ + q] = l;
    }
    __syncthreads();

    if (tid < 16) {
        int r = tid;
        float dd = 0.f;
        #pragma unroll 8
        for (int p = 0; p < P_; ++p) dd = fmaf(yr[r * P_ + p], xr[r * P_ + p], dd);
        dvec[row0 + r] = 0.5f * dd;
    }
}

// ---------------------------------------------------------------------------
// Kernel 2: H = Y @ X^T (split-bf16, 3 WMMA products), fused exp epilogue.
// Block tile 128(M) x 128(N), 8 waves of 64x32 each (mt=4, nt=2), K=128 in LDS.
// 139KB LDS => 1 workgroup per WGP, so launch_bounds frees the full VGPR file.
// ---------------------------------------------------------------------------
__global__ void __launch_bounds__(256, 1)
graph_kernel(const unsigned short* __restrict__ yh, const unsigned short* __restrict__ yl,
             const unsigned short* __restrict__ xh, const unsigned short* __restrict__ xl,
             const float* __restrict__ dvec, float* __restrict__ out)
{
    extern __shared__ char dynsmem[];
    unsigned int* Ah = (unsigned int*)dynsmem;        // 128 rows * 68 u32 each
    unsigned int* Al = Ah + 128 * LDSTRIDE;
    unsigned int* Bh = Al + 128 * LDSTRIDE;
    unsigned int* Bl = Bh + 128 * LDSTRIDE;           // total 4 * 34,816 B = 139,264 B

    const int b    = blockIdx.z;
    const int i0   = blockIdx.y * 128;
    const int j0   = blockIdx.x * 128;
    const int tid  = threadIdx.x;
    const int lane = tid & 31;
    const int wave = tid >> 5;
    const int wMi  = (wave & 1) * 64;    // 2 waves along M
    const int wNj  = (wave >> 1) * 32;   // 4 waves along N
    const int l15  = lane & 15;
    const int sel  = lane >> 4;          // half-wave select

    // --- stage A (y rows) and B (x rows), hi & lo, as b128 chunks ---
    const uint4* gAh = (const uint4*)(yh + (size_t)(b * GS_ + i0) * P_);
    const uint4* gAl = (const uint4*)(yl + (size_t)(b * GS_ + i0) * P_);
    const uint4* gBh = (const uint4*)(xh + (size_t)(b * GS_ + j0) * P_);
    const uint4* gBl = (const uint4*)(xl + (size_t)(b * GS_ + j0) * P_);
    for (int idx = tid; idx < 128 * 16; idx += 256) {   // 16 uint4 per 128-bf16 row
        int r = idx >> 4, c = idx & 15;
        unsigned int off = r * LDSTRIDE + c * 4;        // u32 units, 16B aligned
#ifdef HAVE_ASYNC_LDS
        GASYNC_B128(gAh + idx, Ah + off);
        GASYNC_B128(gAl + idx, Al + off);
        GASYNC_B128(gBh + idx, Bh + off);
        GASYNC_B128(gBl + idx, Bl + off);
#else
        *(uint4*)(Ah + off) = gAh[idx];
        *(uint4*)(Al + off) = gAl[idx];
        *(uint4*)(Bh + off) = gBh[idx];
        *(uint4*)(Bl + off) = gBl[idx];
#endif
    }
#ifdef HAVE_ASYNC_LDS
#if __has_builtin(__builtin_amdgcn_s_wait_asynccnt)
    __builtin_amdgcn_s_wait_asynccnt(0);
#else
    asm volatile("s_wait_asynccnt 0" ::: "memory");
#endif
#endif
    __syncthreads();

    v8f acc[4][2] = {};

    #pragma unroll
    for (int ks = 0; ks < 4; ++ks) {
        const int kw = ks * 16;   // u32 word base of this K=32 slab

        // B fragment (16-bit B 32x16): lane n = x-row j+n; lane<16 -> K0..15, lane>=16 -> K16..31
        Frag bfh[2], bfl[2];
        #pragma unroll
        for (int nt = 0; nt < 2; ++nt) {
            int row = wNj + nt * 16 + l15;
            const unsigned int* ph = Bh + row * LDSTRIDE + kw + sel * 8;
            const unsigned int* pl = Bl + row * LDSTRIDE + kw + sel * 8;
            bfh[nt].q[0] = *(const uint4*)(ph);        // DS_LOAD_B128
            bfh[nt].q[1] = *(const uint4*)(ph + 4);
            bfl[nt].q[0] = *(const uint4*)(pl);
            bfl[nt].q[1] = *(const uint4*)(pl + 4);
        }

        // Per mt: load only this row-tile's A hi/lo, fire its 6 WMMAs, retire.
        // Keeps peak liveness ~115 VGPRs (64 acc + 32 B + 16 A) -> no spills.
        #pragma unroll
        for (int mt = 0; mt < 4; ++mt) {
            // A fragment (16-bit A 16x32): lane<16 -> K0..15, lane>=16 -> K16..31
            Frag ah, al;
            int row = wMi + mt * 16 + l15;
            const unsigned int* ph = Ah + row * LDSTRIDE + kw + sel * 8;
            const unsigned int* pl = Al + row * LDSTRIDE + kw + sel * 8;
            ah.q[0] = *(const uint4*)(ph);
            ah.q[1] = *(const uint4*)(ph + 4);
            al.q[0] = *(const uint4*)(pl);
            al.q[1] = *(const uint4*)(pl + 4);

            // split-bf16: Yh*Xh + Yh*Xl + Yl*Xh; nt-interleave separates
            // successive updates of the same accumulator
            #pragma unroll
            for (int nt = 0; nt < 2; ++nt)
                acc[mt][nt] = __builtin_amdgcn_wmma_f32_16x16x32_bf16(
                    false, ah.v, false, bfh[nt].v, (short)0, acc[mt][nt], false, false);
            #pragma unroll
            for (int nt = 0; nt < 2; ++nt)
                acc[mt][nt] = __builtin_amdgcn_wmma_f32_16x16x32_bf16(
                    false, ah.v, false, bfl[nt].v, (short)0, acc[mt][nt], false, false);
            #pragma unroll
            for (int nt = 0; nt < 2; ++nt)
                acc[mt][nt] = __builtin_amdgcn_wmma_f32_16x16x32_bf16(
                    false, al.v, false, bfh[nt].v, (short)0, acc[mt][nt], false, false);
        }
    }

    // --- fused epilogue: out = exp(-(d_i + d_j - H)) ---
    // C/D layout: VGPR r -> M = r + (lane>=16 ? 8 : 0), N = lane&15
    const float* db = dvec + b * GS_;
    #pragma unroll
    for (int mt = 0; mt < 4; ++mt) {
        int giBase = i0 + wMi + mt * 16 + sel * 8;
        #pragma unroll
        for (int nt = 0; nt < 2; ++nt) {
            int gj = j0 + wNj + nt * 16 + l15;
            float dj = db[gj];
            #pragma unroll
            for (int r = 0; r < 8; ++r) {
                int gi = giBase + r;
                float di = db[gi];
                out[(size_t)(b * GS_ + gi) * OUTW + gj] = __expf(-(di + dj - acc[mt][nt][r]));
            }
        }
    }
}

// ---------------------------------------------------------------------------
// Kernel 3: out[b,i,1024:1152] = x[b,i,:]   (float4 copy)
// ---------------------------------------------------------------------------
__global__ void parax_kernel(const float* __restrict__ x, float* __restrict__ out)
{
    size_t idx = (size_t)blockIdx.x * blockDim.x + threadIdx.x;   // B*GS*32 float4's
    size_t row = idx >> 5;
    int    c   = (int)(idx & 31);
    if (row < (size_t)B_ * GS_) {
        float4 v = ((const float4*)x)[idx];
        ((float4*)(out + row * OUTW + GS_))[c] = v;
    }
}

extern "C" void kernel_launch(void* const* d_in, const int* in_sizes, int n_in,
                              void* d_out, int out_size, void* d_ws, size_t ws_size,
                              hipStream_t stream)
{
    const float* x      = (const float*)d_in[0];
    const float* metrik = (const float*)d_in[1];
    float*       out    = (float*)d_out;

    const size_t n = (size_t)B_ * GS_ * P_;            // 4,194,304 elements
    unsigned short* yh = (unsigned short*)d_ws;        // ws layout: 4 bf16 buffers + d
    unsigned short* yl = yh + n;
    unsigned short* xh = yl + n;
    unsigned short* xl = xh + n;
    float*          dv = (float*)(xl + n);             // total ~32.1 MB of ws

    size_t shm1 = (size_t)(P_ * P_ + 2 * 16 * P_) * sizeof(float);   // 80 KB
    prep_kernel<<<dim3(B_ * GS_ / 16), 256, shm1, stream>>>(x, metrik, yh, yl, xh, xl, dv);

    size_t shm2 = (size_t)(4 * 128) * LDSTRIDE * sizeof(unsigned int); // 139,264 B (< 320KB WGP LDS)
    graph_kernel<<<dim3(GS_ / 128, GS_ / 128, B_), 256, shm2, stream>>>(yh, yl, xh, xl, dv, out);

    parax_kernel<<<dim3((unsigned)((B_ * GS_ * (P_ / 4) + 255) / 256)), 256, 0, stream>>>(x, out);
}
